// TinyGPT_489626271984
// MI455X (gfx1250) — compile-verified, matrix operand order checked by reading, not run on hardware
//
#include <hip/hip_runtime.h>
#include <cstddef>

// ---------------- problem constants ----------------
constexpr int Vv  = 32000;
constexpr int Ll  = 4;
constexpr int Hh  = 4;
constexpr int Dd  = 256;
constexpr int DFf = 1024;
constexpr int Bb  = 2;
constexpr int Tt  = 2048;
constexpr int DHh = 64;          // Dd / Hh
constexpr int Nt  = Bb * Tt;     // 4096 tokens

// ---------------- WMMA types ----------------
typedef __attribute__((ext_vector_type(16))) __bf16 v16bf;
typedef __attribute__((ext_vector_type(8)))  __bf16 bf16x8;
typedef __attribute__((ext_vector_type(8)))  float  v8f;

__device__ __forceinline__ v8f wmma_bf16(v16bf a, v16bf b, v8f c) {
  return __builtin_amdgcn_wmma_f32_16x16x32_bf16(false, a, false, b, (short)0, c,
                                                 false, false);
}

// Load a 16-half A/B fragment slice for one lane:
// halves 0..7  = p[0..7]   (K = kbase+0..7)
// halves 8..15 = p[16..23] (K = kbase+16..23)
__device__ __forceinline__ v16bf ld_frag(const __bf16* __restrict__ p) {
  bf16x8 lo = *reinterpret_cast<const bf16x8*>(p);
  bf16x8 hi = *reinterpret_cast<const bf16x8*>(p + 16);
  v16bf a;
#pragma unroll
  for (int i = 0; i < 8; ++i) { a[i] = lo[i]; a[i + 8] = hi[i]; }
  return a;
}

// ---------------- embedding ----------------
__global__ __launch_bounds__(256) void embed_kernel(
    const int* __restrict__ idx, const float* __restrict__ tok,
    const float* __restrict__ pos, float* __restrict__ x) {
  size_t i = (size_t)blockIdx.x * 256 + threadIdx.x;   // < Nt*Dd
  int n = (int)(i >> 8);
  int d = (int)(i & 255);
  int t = n & (Tt - 1);
  x[i] = tok[(size_t)idx[n] * Dd + d] + pos[(size_t)t * Dd + d];
}

// ---------------- weight transpose + bf16 convert: out[n][k] = in[k][n] ------
__global__ __launch_bounds__(256) void wconv_kernel(
    const float* __restrict__ in, __bf16* __restrict__ out, int Nn, int K) {
  size_t i = (size_t)blockIdx.x * 256 + threadIdx.x;
  if (i >= (size_t)Nn * K) return;
  int n = (int)(i / K);
  int k = (int)(i % K);
  out[i] = (__bf16)in[(size_t)k * Nn + n];
}

// ---------------- layernorm (one row of 256 per block) -> bf16 ---------------
__global__ __launch_bounds__(256) void ln_kernel(
    const float* __restrict__ x, const float* __restrict__ g,
    const float* __restrict__ b, __bf16* __restrict__ out) {
  __shared__ float red[256];
  int row = blockIdx.x, t = threadIdx.x;
  float v = x[(size_t)row * Dd + t];
  red[t] = v;
  __syncthreads();
  for (int s = 128; s > 0; s >>= 1) {
    if (t < s) red[t] += red[t + s];
    __syncthreads();
  }
  float mu = red[0] * (1.0f / 256.0f);
  __syncthreads();
  float d = v - mu;
  red[t] = d * d;
  __syncthreads();
  for (int s = 128; s > 0; s >>= 1) {
    if (t < s) red[t] += red[t + s];
    __syncthreads();
  }
  float var = red[0] * (1.0f / 256.0f);
  float y = d * rsqrtf(var + 1e-5f) * g[t] + b[t];
  out[(size_t)row * Dd + t] = (__bf16)y;
}

// ---------------- generic WMMA GEMM: C[M,Nn] = A[M,K] * Wt[Nn,K]^T + bias ----
// MODE 0: f32 non-temporal store (huge head output, bypass L2 retention)
// MODE 1: f32 residual-add (out = resid + acc)
// MODE 3: GELU -> bf16
// MODE 4: bf16 store head-major [B,H,T,DH]
// MODE 5: bf16 store head-major transposed [B,H,DH,T]
template <int MODE>
__global__ __launch_bounds__(256) void gemm_kernel(
    const __bf16* __restrict__ A, const __bf16* __restrict__ Wt,
    const float* __restrict__ bias, const float* __restrict__ resid,
    float* __restrict__ outF, __bf16* __restrict__ outB,
    int M, int Nn, int K) {
  int w    = threadIdx.x >> 5;
  int lane = threadIdx.x & 31;
  int r    = lane & 15;
  int kb8  = (lane >> 4) * 8;        // 0 or 8
  int m0   = blockIdx.x * 32 + (w >> 2) * 16;
  int n0   = blockIdx.y * 256 + (w & 3) * 64;

  v8f c0 = {}, c1 = {}, c2 = {}, c3 = {};
  const __bf16* Arow = A + (size_t)(m0 + r) * K;
  const __bf16* B0   = Wt + (size_t)(n0 + r) * K;
  const __bf16* B1   = Wt + (size_t)(n0 + 16 + r) * K;
  const __bf16* B2   = Wt + (size_t)(n0 + 32 + r) * K;
  const __bf16* B3   = Wt + (size_t)(n0 + 48 + r) * K;

  // Branch-free body: let the compiler ×2-unroll and software-pipeline the
  // loads into s_clause bursts with partial s_wait_loadcnt ahead of each WMMA.
#pragma unroll 2
  for (int k0 = 0; k0 < K; k0 += 32) {
    v16bf a = ld_frag(Arow + k0 + kb8);
    c0 = wmma_bf16(a, ld_frag(B0 + k0 + kb8), c0);
    c1 = wmma_bf16(a, ld_frag(B1 + k0 + kb8), c1);
    c2 = wmma_bf16(a, ld_frag(B2 + k0 + kb8), c2);
    c3 = wmma_bf16(a, ld_frag(B3 + k0 + kb8), c3);
  }

#pragma unroll
  for (int g = 0; g < 8; ++g) {
    int m = m0 + g + (kb8 ? 8 : 0);
    float vals[4] = {c0[g], c1[g], c2[g], c3[g]};
#pragma unroll
    for (int t2 = 0; t2 < 4; ++t2) {
      int n = n0 + t2 * 16 + r;
      float acc = vals[t2] + bias[n];
      if constexpr (MODE == 0) {
        __builtin_nontemporal_store(acc, &outF[(size_t)m * Nn + n]);
      } else if constexpr (MODE == 1) {
        size_t o = (size_t)m * Nn + n;
        outF[o] = resid[o] + acc;
      } else if constexpr (MODE == 3) {
        float ge = 0.5f * acc * (1.0f + erff(acc * 0.70710678f));
        outB[(size_t)m * Nn + n] = (__bf16)ge;
      } else if constexpr (MODE == 4) {
        int hh = n >> 6, dh = n & 63, bb = m >> 11, tt = m & (Tt - 1);
        outB[(((size_t)bb * Hh + hh) * Tt + tt) * DHh + dh] = (__bf16)acc;
      } else {  // MODE 5
        int hh = n >> 6, dh = n & 63, bb = m >> 11, tt = m & (Tt - 1);
        outB[(((size_t)bb * Hh + hh) * DHh + dh) * Tt + tt] = (__bf16)acc;
      }
    }
  }
}

// ---------------- flash attention: one wave per (b,h,16-query tile) ----------
__global__ __launch_bounds__(32) void attn_kernel(
    const __bf16* __restrict__ qb, const __bf16* __restrict__ kb,
    const __bf16* __restrict__ vtb, __bf16* __restrict__ yb) {
  const int ntile = Tt / 16;                 // 128
  int bid   = blockIdx.x;
  int bh    = bid / ntile;
  int qt    = bid % ntile;
  int h     = bh % Hh;
  int b     = bh / Hh;
  int qbase = qt * 16;
  int lane  = threadIdx.x & 31;
  int r     = lane & 15;
  int kb8   = (lane >> 4) * 8;               // 0 or 8
  int half8 = kb8 ? 8 : 0;

  const __bf16* Q  = qb  + (size_t)bh * Tt * DHh;
  const __bf16* Km = kb  + (size_t)bh * Tt * DHh;
  const __bf16* Vt = vtb + (size_t)bh * DHh * Tt;

  v16bf aq0 = ld_frag(Q + (size_t)(qbase + r) * DHh + kb8);        // dh 0..31
  v16bf aq1 = ld_frag(Q + (size_t)(qbase + r) * DHh + 32 + kb8);   // dh 32..63

  v8f o0 = {}, o1 = {}, o2 = {}, o3 = {};
  float mrow[8], lrow[8];
#pragma unroll
  for (int g = 0; g < 8; ++g) { mrow[g] = -1e30f; lrow[g] = 0.0f; }

  __shared__ __attribute__((aligned(16))) __bf16 Pl[16 * 32];

  for (int kc = 0; kc < qbase + 16; kc += 32) {
    // ---- scores S[16q x 32keys] (two 16-key n-tiles) ----
    v8f s0 = {}, s1 = {};
    {
      int key0 = kc + r;
      s0 = wmma_bf16(aq0, ld_frag(Km + (size_t)key0 * DHh + kb8), s0);
      s0 = wmma_bf16(aq1, ld_frag(Km + (size_t)key0 * DHh + 32 + kb8), s0);
      int key1 = kc + 16 + r;
      s1 = wmma_bf16(aq0, ld_frag(Km + (size_t)key1 * DHh + kb8), s1);
      s1 = wmma_bf16(aq1, ld_frag(Km + (size_t)key1 * DHh + 32 + kb8), s1);
    }
    int key0 = kc + r, key1 = kc + 16 + r;
    float p0[8], p1[8];
#pragma unroll
    for (int g = 0; g < 8; ++g) {
      int qa   = qbase + g + half8;
      float v0 = s0[g] * 0.125f;  if (key0 > qa) v0 = -1e30f;
      float v1 = s1[g] * 0.125f;  if (key1 > qa) v1 = -1e30f;
      float cmax = fmaxf(v0, v1);
#pragma unroll
      for (int msk = 1; msk <= 8; msk <<= 1)
        cmax = fmaxf(cmax, __shfl_xor(cmax, msk, 32));
      float nm    = fmaxf(mrow[g], cmax);
      float alpha = __expf(mrow[g] - nm);
      mrow[g] = nm;
      float e0 = __expf(v0 - nm), e1 = __expf(v1 - nm);
      float rs = e0 + e1;
#pragma unroll
      for (int msk = 1; msk <= 8; msk <<= 1)
        rs += __shfl_xor(rs, msk, 32);
      lrow[g] = lrow[g] * alpha + rs;
      o0[g] *= alpha; o1[g] *= alpha; o2[g] *= alpha; o3[g] *= alpha;
      p0[g] = e0; p1[g] = e1;
    }
    // ---- bounce P through LDS: C-layout -> A-fragment layout ----
    __syncthreads();
#pragma unroll
    for (int g = 0; g < 8; ++g) {
      int row = g + half8;
      Pl[row * 32 + r]      = (__bf16)p0[g];
      Pl[row * 32 + 16 + r] = (__bf16)p1[g];
    }
    __syncthreads();
    v16bf ap;
#pragma unroll
    for (int j = 0; j < 8; ++j) {
      ap[j]     = Pl[r * 32 + kb8 + j];
      ap[j + 8] = Pl[r * 32 + kb8 + 16 + j];
    }
    // ---- O += P @ V  (V pre-transposed: Vt[dh][t], contiguous over keys) ----
    o0 = wmma_bf16(ap, ld_frag(Vt + (size_t)(0  + r) * Tt + kc + kb8), o0);
    o1 = wmma_bf16(ap, ld_frag(Vt + (size_t)(16 + r) * Tt + kc + kb8), o1);
    o2 = wmma_bf16(ap, ld_frag(Vt + (size_t)(32 + r) * Tt + kc + kb8), o2);
    o3 = wmma_bf16(ap, ld_frag(Vt + (size_t)(48 + r) * Tt + kc + kb8), o3);
  }

#pragma unroll
  for (int g = 0; g < 8; ++g) {
    float inv = 1.0f / lrow[g];
    size_t tok = (size_t)b * Tt + qbase + g + half8;
    __bf16* yr = yb + tok * Dd + h * DHh;
    yr[0  + r] = (__bf16)(o0[g] * inv);
    yr[16 + r] = (__bf16)(o1[g] * inv);
    yr[32 + r] = (__bf16)(o2[g] * inv);
    yr[48 + r] = (__bf16)(o3[g] * inv);
  }
}

// ---------------- host-side GEMM dispatch ----------------
static void run_gemm(int mode, const __bf16* A, const __bf16* Wt,
                     const float* bias, const float* resid, float* outF,
                     __bf16* outB, int M, int Nn, int K, hipStream_t s) {
  dim3 grid(M / 32, Nn / 256), block(256);
  switch (mode) {
    case 0: gemm_kernel<0><<<grid, block, 0, s>>>(A, Wt, bias, resid, outF, outB, M, Nn, K); break;
    case 1: gemm_kernel<1><<<grid, block, 0, s>>>(A, Wt, bias, resid, outF, outB, M, Nn, K); break;
    case 3: gemm_kernel<3><<<grid, block, 0, s>>>(A, Wt, bias, resid, outF, outB, M, Nn, K); break;
    case 4: gemm_kernel<4><<<grid, block, 0, s>>>(A, Wt, bias, resid, outF, outB, M, Nn, K); break;
    default: gemm_kernel<5><<<grid, block, 0, s>>>(A, Wt, bias, resid, outF, outB, M, Nn, K); break;
  }
}

static void run_wconv(const float* in, __bf16* out, int Nn, int K, hipStream_t s) {
  size_t tot = (size_t)Nn * K;
  wconv_kernel<<<(unsigned)((tot + 255) / 256), 256, 0, s>>>(in, out, Nn, K);
}

extern "C" void kernel_launch(void* const* d_in, const int* in_sizes, int n_in,
                              void* d_out, int out_size, void* d_ws, size_t ws_size,
                              hipStream_t stream) {
  (void)in_sizes; (void)n_in; (void)out_size;
  const int*   idx   = (const int*)  d_in[0];
  const float* tok   = (const float*)d_in[1];
  const float* pos   = (const float*)d_in[2];
  const float* Wq    = (const float*)d_in[3];
  const float* bq    = (const float*)d_in[4];
  const float* Wk    = (const float*)d_in[5];
  const float* bk    = (const float*)d_in[6];
  const float* Wv    = (const float*)d_in[7];
  const float* bv    = (const float*)d_in[8];
  const float* Wo    = (const float*)d_in[9];
  const float* bo    = (const float*)d_in[10];
  const float* ln1g  = (const float*)d_in[11];
  const float* ln1b  = (const float*)d_in[12];
  const float* W1    = (const float*)d_in[13];
  const float* b1    = (const float*)d_in[14];
  const float* W2    = (const float*)d_in[15];
  const float* b2    = (const float*)d_in[16];
  const float* ln2g  = (const float*)d_in[17];
  const float* ln2b  = (const float*)d_in[18];
  const float* lnfg  = (const float*)d_in[19];
  const float* lnfb  = (const float*)d_in[20];
  const float* headW = (const float*)d_in[21];
  const float* headb = (const float*)d_in[22];
  float* out = (float*)d_out;

  char* ws = (char*)d_ws;
  size_t off = 0;
  auto alloc = [&](size_t bytes) -> void* {
    void* p = ws + off;
    off += (bytes + 255) & ~(size_t)255;
    return p;
  };
  float*  x    = (float*) alloc((size_t)Nt * Dd * 4);
  __bf16* hb   = (__bf16*)alloc((size_t)Nt * Dd * 2);
  __bf16* qbuf = (__bf16*)alloc((size_t)Nt * Dd * 2);
  __bf16* kbuf = (__bf16*)alloc((size_t)Nt * Dd * 2);
  __bf16* vtb  = (__bf16*)alloc((size_t)Nt * Dd * 2);
  __bf16* ybuf = (__bf16*)alloc((size_t)Nt * Dd * 2);
  __bf16* g1   = (__bf16*)alloc((size_t)Nt * DFf * 2);
  __bf16* wqT  = (__bf16*)alloc((size_t)Dd * Dd * 2);
  __bf16* wkT  = (__bf16*)alloc((size_t)Dd * Dd * 2);
  __bf16* wvT  = (__bf16*)alloc((size_t)Dd * Dd * 2);
  __bf16* woT  = (__bf16*)alloc((size_t)Dd * Dd * 2);
  __bf16* w1T  = (__bf16*)alloc((size_t)Dd * DFf * 2);
  __bf16* w2T  = (__bf16*)alloc((size_t)Dd * DFf * 2);
  __bf16* whT  = (__bf16*)alloc((size_t)Vv * Dd * 2);
  if (off > ws_size) return;  // insufficient workspace

  embed_kernel<<<Nt * Dd / 256, 256, 0, stream>>>(idx, tok, pos, x);
  run_wconv(headW, whT, Vv, Dd, stream);  // head_W [D,V] -> whT [V,D]

  for (int l = 0; l < Ll; ++l) {
    ln_kernel<<<Nt, 256, 0, stream>>>(x, ln1g + l * Dd, ln1b + l * Dd, hb);
    run_wconv(Wq + (size_t)l * Dd * Dd, wqT, Dd, Dd, stream);
    run_wconv(Wk + (size_t)l * Dd * Dd, wkT, Dd, Dd, stream);
    run_wconv(Wv + (size_t)l * Dd * Dd, wvT, Dd, Dd, stream);
    run_wconv(Wo + (size_t)l * Dd * Dd, woT, Dd, Dd, stream);

    run_gemm(4, hb, wqT, bq + l * Dd, nullptr, nullptr, qbuf, Nt, Dd, Dd, stream);
    run_gemm(4, hb, wkT, bk + l * Dd, nullptr, nullptr, kbuf, Nt, Dd, Dd, stream);
    run_gemm(5, hb, wvT, bv + l * Dd, nullptr, nullptr, vtb,  Nt, Dd, Dd, stream);

    attn_kernel<<<Bb * Hh * (Tt / 16), 32, 0, stream>>>(qbuf, kbuf, vtb, ybuf);

    run_gemm(1, ybuf, woT, bo + l * Dd, x, x, nullptr, Nt, Dd, Dd, stream);

    ln_kernel<<<Nt, 256, 0, stream>>>(x, ln2g + l * Dd, ln2b + l * Dd, hb);
    run_wconv(W1 + (size_t)l * Dd * DFf, w1T, DFf, Dd, stream);  // [D,DF]->[DF,D]
    run_wconv(W2 + (size_t)l * DFf * Dd, w2T, Dd, DFf, stream);  // [DF,D]->[D,DF]
    run_gemm(3, hb, w1T, b1 + l * DFf, nullptr, nullptr, g1, Nt, DFf, Dd, stream);
    run_gemm(1, g1, w2T, b2 + l * Dd, x, x, nullptr, Nt, Dd, DFf, stream);
  }

  ln_kernel<<<Nt, 256, 0, stream>>>(x, lnfg, lnfb, hb);
  run_gemm(0, hb, whT, headb, nullptr, out, nullptr, Nt, Vv, Dd, stream);
}
